// TransformerDecoderBlock_38920993636931
// MI455X (gfx1250) — matrix operand order repaired and checked by reading, not compiled, not run on hardware
//
#include <hip/hip_runtime.h>

// ---------------- problem constants ----------------
#define HH 12
#define EE 768
#define HD 64
#define BB 8
#define SS 512
#define LL 512
// tokens = BB*SS = 4096

typedef unsigned short u16;
typedef __bf16 bf16t;
typedef __attribute__((ext_vector_type(16))) bf16t v16bf;
typedef __attribute__((ext_vector_type(8)))  float v8f;

union FragBF { v16bf v; int4 q[2]; };

__device__ static inline u16 f2bf(float f) {
  unsigned int u = __float_as_uint(f);
  unsigned int r = (u + 0x7FFFu + ((u >> 16) & 1u)) >> 16;
  return (u16)r;
}

__device__ static inline v8f wmma_bf16(v16bf a, v16bf b, v8f c) {
  // D = A(16x32 bf16) * B(32x16 bf16) + C(16x16 f32)
  return __builtin_amdgcn_wmma_f32_16x16x32_bf16(false, a, false, b, (short)0, c,
                                                 false, false);
}

__device__ static inline float gelu_f(float x) {
  return 0.5f * x * (1.0f + erff(x * 0.70710678118654752440f));
}

// ---- CDNA5 async memory->LDS copy (16B per lane), tracked by ASYNCcnt ----
// LDS address operand = low 32 bits of the flat address (ISA 10.2: for the LDS
// aperture, the low 32 bits are the LDS offset).
__device__ static inline void async_ld16(const void* g, void* lds) {
  unsigned l = (unsigned)(size_t)lds;
  asm volatile("global_load_async_to_lds_b128 %0, %1, off"
               :: "v"(l), "v"(g) : "memory");
}
__device__ static inline void wait_async_le4() {
  asm volatile("s_wait_asynccnt 0x4" ::: "memory");
}
__device__ static inline void wait_async_0() {
  asm volatile("s_wait_asynccnt 0x0" ::: "memory");
}

// ---------------- elementwise prep kernels ----------------

__global__ __launch_bounds__(256) void cast_bf16_kernel(const float* __restrict__ in,
                                                        u16* __restrict__ out, int n) {
  int i = blockIdx.x * 256 + threadIdx.x;
  if (i < n) out[i] = f2bf(in[i]);
}

// w[H, E, HD] -> Wt[(h*HD+d)][e]  (bf16, [N][K] with N=H*HD, K=E)
__global__ __launch_bounds__(256) void wtrans_hed_kernel(const float* __restrict__ in,
                                                         u16* __restrict__ out) {
  int i = blockIdx.x * 256 + threadIdx.x;
  if (i >= EE * EE) return;
  int e = i % EE;
  int n = i / EE;
  int h = n >> 6;       // / HD
  int d = n & 63;       // % HD
  out[i] = f2bf(in[((size_t)h * EE + e) * HD + d]);
}

// w[K, N] -> Wt[n][k] (bf16)
__global__ __launch_bounds__(256) void wtrans_kn_kernel(const float* __restrict__ in,
                                                        u16* __restrict__ out) {
  int i = blockIdx.x * 256 + threadIdx.x;
  if (i >= EE * EE) return;
  int k = i % EE;
  int n = i / EE;
  out[i] = f2bf(in[(size_t)k * EE + n]);
}

// ---------------- WMMA GEMM: C[M,N] = A[M,K] @ Bt[N,K]^T ----------------
// Block tile 128x128, BK=32, 8 waves (2x4), wave tile 64x32 (4x2 WMMA tiles).
// Double-buffered LDS tiles filled by async global->LDS copies so tile t+1's
// HBM traffic overlaps tile t's WMMAs.
// EPI: 0 = store bf16; 1 = +bias, store f32; 2 = +bias, gelu, store bf16.
template <int EPI>
__global__ __launch_bounds__(256) void gemm_bf16_kernel(
    const u16* __restrict__ A, const u16* __restrict__ Bt,
    const float* __restrict__ bias, float* __restrict__ Cf, u16* __restrict__ Cb,
    int M, int N, int K) {
  __shared__ u16 la[2][128][40];  // A tile  [128 rows][32 K] (pitch 40 -> 16B aligned)
  __shared__ u16 lb[2][128][40];  // Bt tile [128 cols][32 K]

  const int t = threadIdx.x;
  const int m0 = blockIdx.y * 128, n0 = blockIdx.x * 128;
  const int wid = t >> 5, lane = t & 31;
  const int r = lane & 15, kh = lane >> 4;
  const int wm = wid & 1, wn = wid >> 1;

  v8f acc[4][2];
  const v8f z8 = {0.f, 0.f, 0.f, 0.f, 0.f, 0.f, 0.f, 0.f};
#pragma unroll
  for (int i = 0; i < 4; ++i)
#pragma unroll
    for (int j = 0; j < 2; ++j) acc[i][j] = z8;

  // issue async staging of one K tile into LDS buffer `buf`
  auto stage = [&](int buf, int k0) {
    int c = t;
#pragma unroll
    for (int pass = 0; pass < 2; ++pass, c += 256) {
      int row = c >> 2, col8 = (c & 3) << 3;
      async_ld16(&A[(size_t)(m0 + row) * K + k0 + col8], &la[buf][row][col8]);
      async_ld16(&Bt[(size_t)(n0 + row) * K + k0 + col8], &lb[buf][row][col8]);
    }
  };

  const int nIter = K / 32;
  stage(0, 0);
  for (int it = 0; it < nIter; ++it) {
    const int buf = it & 1;
    __syncthreads();  // all waves finished reading buf^1 (tile it-1)
    if (it + 1 < nIter) {
      stage(buf ^ 1, (it + 1) * 32);  // overlap next tile with this compute
      wait_async_le4();               // in-order: tile `it`'s 4 loads complete
    } else {
      wait_async_0();
    }
    __syncthreads();  // tile `it` visible to all waves

    FragBF af[4], bfr[2];
#pragma unroll
    for (int i = 0; i < 4; ++i) {
      int row = wm * 64 + i * 16 + r;
      af[i].q[0] = *(const int4*)&la[buf][row][kh * 8];
      af[i].q[1] = *(const int4*)&la[buf][row][16 + kh * 8];
    }
#pragma unroll
    for (int j = 0; j < 2; ++j) {
      int col = wn * 32 + j * 16 + r;
      bfr[j].q[0] = *(const int4*)&lb[buf][col][kh * 8];
      bfr[j].q[1] = *(const int4*)&lb[buf][col][16 + kh * 8];
    }
#pragma unroll
    for (int i = 0; i < 4; ++i)
#pragma unroll
      for (int j = 0; j < 2; ++j)
        acc[i][j] = wmma_bf16(af[i].v, bfr[j].v, acc[i][j]);
  }

  // epilogue: acc element p of lane -> row offset p + kh*8, col = lane%16
#pragma unroll
  for (int j = 0; j < 2; ++j) {
    int col = n0 + wn * 32 + j * 16 + r;
    float bv = (EPI != 0) ? bias[col] : 0.f;
#pragma unroll
    for (int i = 0; i < 4; ++i) {
#pragma unroll
      for (int p = 0; p < 8; ++p) {
        int row = m0 + wm * 64 + i * 16 + p + kh * 8;
        float v = acc[i][j][p] + bv;
        if (EPI == 2) v = gelu_f(v);
        if (EPI == 1)
          Cf[(size_t)row * N + col] = v;
        else
          Cb[(size_t)row * N + col] = f2bf(v);
      }
    }
  }
}

// ---------------- flash attention (values == kx, mask like reference) ----------
// OUT[bq, h*64+d] = softmax( (QX@KX^T)/8 masked ) @ KX, per (b,h)
// grid: (BB*HH, Sq/128), 256 threads = 8 waves, each wave owns 16 q-rows.
__global__ __launch_bounds__(256) void attn_kernel(
    const u16* __restrict__ QX, const u16* __restrict__ KX,
    const int* __restrict__ vlens,  // nullptr -> causal mask
    u16* __restrict__ OUT, int Sq, int KL) {
  __shared__ u16 kt[32][72];       // K tile row-major [32 keys][64 d]
  __shared__ u16 vt[64][40];       // K tile transposed [64 d][32 keys]
  __shared__ u16 pb[8][16][40];    // per-wave P relayout buffer

  const int t = threadIdx.x;
  const int bh = blockIdx.x;
  const int b = bh / HH, h = bh % HH;
  const int wid = t >> 5, lane = t & 31;
  const int r = lane & 15, kh = lane >> 4;
  const int qw = blockIdx.y * 128 + wid * 16;
  const bool causal = (vlens == nullptr);
  const int vlen = causal ? 0 : vlens[b];

  // load Q fragments (A-frags, K = d dimension, 2 chunks of 32)
  const u16* qbase = QX + ((size_t)(b * Sq + qw + r)) * EE + h * HD;
  FragBF aq[2];
  aq[0].q[0] = *(const int4*)&qbase[kh * 8];
  aq[0].q[1] = *(const int4*)&qbase[16 + kh * 8];
  aq[1].q[0] = *(const int4*)&qbase[32 + kh * 8];
  aq[1].q[1] = *(const int4*)&qbase[48 + kh * 8];

  const v8f z8 = {0.f, 0.f, 0.f, 0.f, 0.f, 0.f, 0.f, 0.f};
  v8f o0 = z8, o1 = z8, o2 = z8, o3 = z8;
  float mrow[8], srow[8];
#pragma unroll
  for (int p = 0; p < 8; ++p) { mrow[p] = -3.0e38f; srow[p] = 0.f; }

  for (int k0 = 0; k0 < KL; k0 += 32) {
    // stage 32 keys x 64 dims (row-major and transposed)
    {
      int kk = t >> 3, d0 = (t & 7) << 3;
      int4 ch = *(const int4*)&KX[((size_t)(b * KL + k0 + kk)) * EE + h * HD + d0];
      *(int4*)&kt[kk][d0] = ch;
      const u16* cs = (const u16*)&ch;
#pragma unroll
      for (int i = 0; i < 8; ++i) vt[d0 + i][kk] = cs[i];
    }
    __syncthreads();

    // scores S[16 x 32] = Q[16x64] @ K_tile^T : two 16x16 accumulators
    FragBF bk[2][2];
#pragma unroll
    for (int j = 0; j < 2; ++j)
#pragma unroll
      for (int kc = 0; kc < 2; ++kc) {
        int c = j * 16 + r;
        bk[j][kc].q[0] = *(const int4*)&kt[c][kc * 32 + kh * 8];
        bk[j][kc].q[1] = *(const int4*)&kt[c][kc * 32 + 16 + kh * 8];
      }
    v8f s0 = wmma_bf16(aq[0].v, bk[0][0].v, z8);
    s0 = wmma_bf16(aq[1].v, bk[0][1].v, s0);
    v8f s1 = wmma_bf16(aq[0].v, bk[1][0].v, z8);
    s1 = wmma_bf16(aq[1].v, bk[1][1].v, s1);

    // scale + mask (same semantics as reference: scale then fill -1e4)
    float x0[8], x1[8], mc[8];
    const int key0 = k0 + r, key1 = k0 + 16 + r;
#pragma unroll
    for (int p = 0; p < 8; ++p) {
      int qg = qw + p + kh * 8;
      float a = s0[p] * 0.125f;
      float c2 = s1[p] * 0.125f;
      if (causal) {
        if (key0 > qg) a = -1e4f;
        if (key1 > qg) c2 = -1e4f;
      } else {
        if (key0 >= vlen) a = -1e4f;
        if (key1 >= vlen) c2 = -1e4f;
      }
      x0[p] = a; x1[p] = c2;
      mc[p] = fmaxf(a, c2);
    }
    // row reductions live in half-wave groups of 16 lanes (xor masks 1..8)
#pragma unroll
    for (int p = 0; p < 8; ++p) {
#pragma unroll
      for (int ofs = 8; ofs > 0; ofs >>= 1)
        mc[p] = fmaxf(mc[p], __shfl_xor(mc[p], ofs, 32));
    }
    float sf[8];
#pragma unroll
    for (int p = 0; p < 8; ++p) {
      float mn = fmaxf(mrow[p], mc[p]);
      sf[p] = __expf(mrow[p] - mn);
      mrow[p] = mn;
      float p0 = __expf(x0[p] - mn);
      float p1 = __expf(x1[p] - mn);
      float rs = p0 + p1;
#pragma unroll
      for (int ofs = 8; ofs > 0; ofs >>= 1) rs += __shfl_xor(rs, ofs, 32);
      srow[p] = srow[p] * sf[p] + rs;
      int rowp = p + kh * 8;
      pb[wid][rowp][r] = f2bf(p0);
      pb[wid][rowp][16 + r] = f2bf(p1);
    }
#pragma unroll
    for (int p = 0; p < 8; ++p) {
      o0[p] *= sf[p]; o1[p] *= sf[p]; o2[p] *= sf[p]; o3[p] *= sf[p];
    }

    // relayout P (accumulator layout) -> A-fragment via wave-private LDS
    FragBF ap;
    ap.q[0] = *(const int4*)&pb[wid][r][kh * 8];
    ap.q[1] = *(const int4*)&pb[wid][r][16 + kh * 8];

    // O[16x64] += P[16x32] @ V[32x64]  (V = K tile), 4 N-subtiles
    {
      FragBF bv;
      int d = r;
      bv.q[0] = *(const int4*)&vt[d][kh * 8];
      bv.q[1] = *(const int4*)&vt[d][16 + kh * 8];
      o0 = wmma_bf16(ap.v, bv.v, o0);
      d = 16 + r;
      bv.q[0] = *(const int4*)&vt[d][kh * 8];
      bv.q[1] = *(const int4*)&vt[d][16 + kh * 8];
      o1 = wmma_bf16(ap.v, bv.v, o1);
      d = 32 + r;
      bv.q[0] = *(const int4*)&vt[d][kh * 8];
      bv.q[1] = *(const int4*)&vt[d][16 + kh * 8];
      o2 = wmma_bf16(ap.v, bv.v, o2);
      d = 48 + r;
      bv.q[0] = *(const int4*)&vt[d][kh * 8];
      bv.q[1] = *(const int4*)&vt[d][16 + kh * 8];
      o3 = wmma_bf16(ap.v, bv.v, o3);
    }
    __syncthreads();
  }

  // normalize and store bf16 [tokens, H*HD]
#pragma unroll
  for (int p = 0; p < 8; ++p) {
    float inv = 1.0f / srow[p];
    size_t rowoff = ((size_t)(b * Sq + qw + p + kh * 8)) * EE + h * HD;
    OUT[rowoff + r]      = f2bf(o0[p] * inv);
    OUT[rowoff + 16 + r] = f2bf(o1[p] * inv);
    OUT[rowoff + 32 + r] = f2bf(o2[p] * inv);
    OUT[rowoff + 48 + r] = f2bf(o3[p] * inv);
  }
}

// ---------------- residual + LayerNorm (f32 out + optional bf16 out) --------
__global__ __launch_bounds__(256) void ln_kernel(const float* __restrict__ A,
                                                 const float* __restrict__ R,
                                                 const float* __restrict__ w,
                                                 const float* __restrict__ bch,
                                                 float* __restrict__ Of,
                                                 u16* __restrict__ Ob) {
  __shared__ float red[8];
  const int row = blockIdx.x, t = threadIdx.x;
  const size_t base = (size_t)row * EE;
  float xv[3];
  float s = 0.f;
#pragma unroll
  for (int i = 0; i < 3; ++i) {
    int c = t + i * 256;
    xv[i] = A[base + c] + R[base + c];
    s += xv[i];
  }
#pragma unroll
  for (int ofs = 16; ofs > 0; ofs >>= 1) s += __shfl_xor(s, ofs, 32);
  if ((t & 31) == 0) red[t >> 5] = s;
  __syncthreads();
  s = 0.f;
#pragma unroll
  for (int i = 0; i < 8; ++i) s += red[i];
  const float mean = s * (1.0f / (float)EE);

  float v = 0.f;
#pragma unroll
  for (int i = 0; i < 3; ++i) {
    float d = xv[i] - mean;
    v += d * d;
  }
  __syncthreads();
#pragma unroll
  for (int ofs = 16; ofs > 0; ofs >>= 1) v += __shfl_xor(v, ofs, 32);
  if ((t & 31) == 0) red[t >> 5] = v;
  __syncthreads();
  v = 0.f;
#pragma unroll
  for (int i = 0; i < 8; ++i) v += red[i];
  const float inv = rsqrtf(v * (1.0f / (float)EE) + 1e-12f);

#pragma unroll
  for (int i = 0; i < 3; ++i) {
    int c = t + i * 256;
    float y = w[c] * (xv[i] - mean) * inv + bch[c];
    Of[base + c] = y;
    if (Ob) Ob[base + c] = f2bf(y);
  }
}

// ---------------- orchestration ----------------
extern "C" void kernel_launch(void* const* d_in, const int* in_sizes, int n_in,
                              void* d_out, int out_size, void* d_ws, size_t ws_size,
                              hipStream_t stream) {
  (void)in_sizes; (void)n_in; (void)out_size; (void)ws_size;

  const float* X    = (const float*)d_in[0];
  const float* ENC  = (const float*)d_in[1];
  const int*   VLEN = (const int*)d_in[2];
  const float* Wk1  = (const float*)d_in[3];
  const float* Wq1  = (const float*)d_in[4];
  const float* P1   = (const float*)d_in[5];
  const float* P1b  = (const float*)d_in[6];
  const float* L1w  = (const float*)d_in[7];
  const float* L1b  = (const float*)d_in[8];
  const float* Wk2  = (const float*)d_in[9];
  const float* Wq2  = (const float*)d_in[10];
  const float* P2   = (const float*)d_in[11];
  const float* P2b  = (const float*)d_in[12];
  const float* L2w  = (const float*)d_in[13];
  const float* L2b  = (const float*)d_in[14];
  const float* F1w  = (const float*)d_in[15];
  const float* F1bi = (const float*)d_in[16];
  const float* F2w  = (const float*)d_in[17];
  const float* F2b  = (const float*)d_in[18];
  const float* L3w  = (const float*)d_in[19];
  const float* L3b  = (const float*)d_in[20];
  float* OUTF = (float*)d_out;

  const size_t TOK = (size_t)BB * SS;       // 4096
  const size_t MAT = TOK * EE;              // 3,145,728 elems
  const size_t WEL = (size_t)EE * EE;       // 589,824 elems

  char* ws = (char*)d_ws;
  size_t off = 0;
  auto take = [&](size_t bytes) -> char* {
    char* p = ws + off;
    off = (off + bytes + 255) & ~(size_t)255;
    return p;
  };

  u16* Xbf  = (u16*)take(MAT * 2);
  u16* Ebf  = (u16*)take(MAT * 2);
  u16* KXb  = (u16*)take(MAT * 2);
  u16* QXb  = (u16*)take(MAT * 2);
  u16* ATb  = (u16*)take(MAT * 2);
  u16* Ybf  = (u16*)take(MAT * 2);
  u16* Zbf  = (u16*)take(MAT * 2);
  u16* F1b  = (u16*)take(MAT * 2);
  u16* Wk1t = (u16*)take(WEL * 2);
  u16* Wq1t = (u16*)take(WEL * 2);
  u16* P1t  = (u16*)take(WEL * 2);
  u16* Wk2t = (u16*)take(WEL * 2);
  u16* Wq2t = (u16*)take(WEL * 2);
  u16* P2t  = (u16*)take(WEL * 2);
  u16* W1t  = (u16*)take(WEL * 2);
  u16* W2t  = (u16*)take(WEL * 2);
  float* Cf = (float*)take(MAT * 4);
  float* Yf = (float*)take(MAT * 4);
  float* Zf = (float*)take(MAT * 4);

  const dim3 blk(256);
  const dim3 gEl((unsigned)((MAT + 255) / 256));
  const dim3 gW((unsigned)((WEL + 255) / 256));
  const dim3 gGemm(EE / 128, (unsigned)(TOK / 128));  // (6, 32)
  const dim3 gAttn(BB * HH, SS / 128);                // (96, 4)
  const dim3 gLN((unsigned)TOK);

  // --- prep: casts + weight transposes ---
  cast_bf16_kernel<<<gEl, blk, 0, stream>>>(X, Xbf, (int)MAT);
  cast_bf16_kernel<<<gEl, blk, 0, stream>>>(ENC, Ebf, (int)MAT);
  wtrans_hed_kernel<<<gW, blk, 0, stream>>>(Wk1, Wk1t);
  wtrans_hed_kernel<<<gW, blk, 0, stream>>>(Wq1, Wq1t);
  wtrans_kn_kernel<<<gW, blk, 0, stream>>>(P1, P1t);
  wtrans_hed_kernel<<<gW, blk, 0, stream>>>(Wk2, Wk2t);
  wtrans_hed_kernel<<<gW, blk, 0, stream>>>(Wq2, Wq2t);
  wtrans_kn_kernel<<<gW, blk, 0, stream>>>(P2, P2t);
  wtrans_kn_kernel<<<gW, blk, 0, stream>>>(F1w, W1t);
  wtrans_kn_kernel<<<gW, blk, 0, stream>>>(F2w, W2t);

  // --- self attention ---
  gemm_bf16_kernel<0><<<gGemm, blk, 0, stream>>>(Xbf, Wk1t, nullptr, nullptr, KXb,
                                                 (int)TOK, EE, EE);
  gemm_bf16_kernel<0><<<gGemm, blk, 0, stream>>>(Xbf, Wq1t, nullptr, nullptr, QXb,
                                                 (int)TOK, EE, EE);
  attn_kernel<<<gAttn, blk, 0, stream>>>(QXb, KXb, nullptr, ATb, SS, SS);
  gemm_bf16_kernel<1><<<gGemm, blk, 0, stream>>>(ATb, P1t, P1b, Cf, nullptr,
                                                 (int)TOK, EE, EE);
  ln_kernel<<<gLN, blk, 0, stream>>>(Cf, X, L1w, L1b, Yf, Ybf);

  // --- cross attention ---
  gemm_bf16_kernel<0><<<gGemm, blk, 0, stream>>>(Ebf, Wk2t, nullptr, nullptr, KXb,
                                                 (int)TOK, EE, EE);
  gemm_bf16_kernel<0><<<gGemm, blk, 0, stream>>>(Ybf, Wq2t, nullptr, nullptr, QXb,
                                                 (int)TOK, EE, EE);
  attn_kernel<<<gAttn, blk, 0, stream>>>(QXb, KXb, VLEN, ATb, SS, LL);
  gemm_bf16_kernel<1><<<gGemm, blk, 0, stream>>>(ATb, P2t, P2b, Cf, nullptr,
                                                 (int)TOK, EE, EE);
  ln_kernel<<<gLN, blk, 0, stream>>>(Cf, Yf, L2w, L2b, Zf, Zbf);

  // --- FFN ---
  gemm_bf16_kernel<2><<<gGemm, blk, 0, stream>>>(Zbf, W1t, F1bi, nullptr, F1b,
                                                 (int)TOK, EE, EE);
  gemm_bf16_kernel<1><<<gGemm, blk, 0, stream>>>(F1b, W2t, F2b, Cf, nullptr,
                                                 (int)TOK, EE, EE);
  ln_kernel<<<gLN, blk, 0, stream>>>(Cf, Zf, L3w, L3b, OUTF, nullptr);
}
